// Autoformer_29351806501044
// MI455X (gfx1250) — matrix operand order repaired and checked by reading, not compiled, not run on hardware
//
#include <hip/hip_runtime.h>
#include <math.h>

typedef __attribute__((ext_vector_type(16))) _Float16 v16h;
typedef __attribute__((ext_vector_type(8)))  _Float16 v8h;
typedef __attribute__((ext_vector_type(8)))  float    v8f;
typedef unsigned int u32x4 __attribute__((ext_vector_type(4)));
typedef int          i32x4 __attribute__((ext_vector_type(4)));
typedef int          i32x8 __attribute__((ext_vector_type(8)));

#define MROWS 16384   // B*L
#define BB    16
#define LL    1024
#define DM    512
#define DFF   2048
#define TOPK  34      // int(5*log(1024))

__device__ __forceinline__ v16h cat8(v8h lo, v8h hi) {
  return __builtin_shufflevector(lo, hi, 0,1,2,3,4,5,6,7,8,9,10,11,12,13,14,15);
}

// ---------------------------------------------------------------------------
// f32 -> f16 conversion / fill
// ---------------------------------------------------------------------------
__global__ void k_cvt(const float* __restrict__ s, _Float16* __restrict__ d, size_t n) {
  size_t i = (size_t)blockIdx.x * 256 + threadIdx.x;
  if (i < n) d[i] = (_Float16)s[i];
}

__global__ void k_fill(float* p, size_t n) {
  size_t i = (size_t)blockIdx.x * 256 + threadIdx.x;
  if (i < n) p[i] = 0.f;
}

// ---------------------------------------------------------------------------
// WMMA GEMM: C[M,N] = act(A[M,K] * W[N,K]^T + bias) (+ residual)
// A,W f16 row-major (K contiguous), C f32 row-major.
// 4 waves/block in 2x2; each wave computes a 32x64 strip (2 A-frags x
// 4 B-frags = 8 v_wmma_f32_16x16x32_f16 per 32-deep K step), with explicit
// register double-buffering so next-step loads overlap current WMMAs.
// ---------------------------------------------------------------------------
struct Frag {
  v8h a0lo, a0hi, a1lo, a1hi;
  v16h b0, b1, b2, b3;
};

__device__ __forceinline__ Frag frag_load(const _Float16* arow0, const _Float16* arow1,
                                          const _Float16* bp0, const _Float16* bp1,
                                          const _Float16* bp2, const _Float16* bp3,
                                          int k0) {
  Frag f;
  f.a0lo = *(const v8h*)(arow0 + k0);
  f.a0hi = *(const v8h*)(arow0 + k0 + 8);
  f.a1lo = *(const v8h*)(arow1 + k0);
  f.a1hi = *(const v8h*)(arow1 + k0 + 8);
  f.b0 = *(const v16h*)(bp0 + k0);
  f.b1 = *(const v16h*)(bp1 + k0);
  f.b2 = *(const v16h*)(bp2 + k0);
  f.b3 = *(const v16h*)(bp3 + k0);
  return f;
}

__device__ __forceinline__ void frag_mma(const Frag& f, v8f (&acc)[2][4]) {
  v16h af0 = cat8(f.a0lo, f.a0hi);
  v16h af1 = cat8(f.a1lo, f.a1hi);
  acc[0][0] = __builtin_amdgcn_wmma_f32_16x16x32_f16(false, af0, false, f.b0, (short)0, acc[0][0], false, false);
  acc[0][1] = __builtin_amdgcn_wmma_f32_16x16x32_f16(false, af0, false, f.b1, (short)0, acc[0][1], false, false);
  acc[0][2] = __builtin_amdgcn_wmma_f32_16x16x32_f16(false, af0, false, f.b2, (short)0, acc[0][2], false, false);
  acc[0][3] = __builtin_amdgcn_wmma_f32_16x16x32_f16(false, af0, false, f.b3, (short)0, acc[0][3], false, false);
  acc[1][0] = __builtin_amdgcn_wmma_f32_16x16x32_f16(false, af1, false, f.b0, (short)0, acc[1][0], false, false);
  acc[1][1] = __builtin_amdgcn_wmma_f32_16x16x32_f16(false, af1, false, f.b1, (short)0, acc[1][1], false, false);
  acc[1][2] = __builtin_amdgcn_wmma_f32_16x16x32_f16(false, af1, false, f.b2, (short)0, acc[1][2], false, false);
  acc[1][3] = __builtin_amdgcn_wmma_f32_16x16x32_f16(false, af1, false, f.b3, (short)0, acc[1][3], false, false);
}

__global__ __launch_bounds__(128)
void k_gemm(const _Float16* __restrict__ A, const _Float16* __restrict__ W,
            const float* __restrict__ bias, const float* __restrict__ resid,
            float* __restrict__ C, int M, int N, int K, int act,
            long long sA, long long sW, long long sC)
{
  const int z = blockIdx.z;
  A += (size_t)z * sA;
  W += (size_t)z * sW;
  C += (size_t)z * sC;
  if (resid) resid += (size_t)z * sC;

  const int wave = threadIdx.x >> 5;
  const int lane = threadIdx.x & 31;
  const int hs   = lane >> 4;      // 0: lanes 0-15, 1: lanes 16-31
  const int lr   = lane & 15;
  const int m0   = blockIdx.y * 64  + (wave >> 1) * 32;
  const int n0   = blockIdx.x * 128 + (wave & 1) * 64;

  v8f acc[2][4] = {};

  // A fragment base (lane row m0+lr, 16 contiguous K halves)
  const _Float16* arow0 = A + (size_t)(m0 + lr) * K + hs * 16;
  const _Float16* arow1 = arow0 + (size_t)16 * K;
  // B fragment base (lane column n0+lr = row of W, 16 contiguous K halves)
  const _Float16* bp0 = W + (size_t)(n0 + lr) * K + hs * 16;
  const _Float16* bp1 = bp0 + (size_t)16 * K;
  const _Float16* bp2 = bp0 + (size_t)32 * K;
  const _Float16* bp3 = bp0 + (size_t)48 * K;

  Frag cur = frag_load(arow0, arow1, bp0, bp1, bp2, bp3, 0);
  for (int k0 = 0; k0 < K - 32; k0 += 32) {
    __builtin_prefetch(arow0 + k0 + 160, 0, 3);
    Frag nxt = frag_load(arow0, arow1, bp0, bp1, bp2, bp3, k0 + 32);
    frag_mma(cur, acc);
    cur = nxt;
  }
  frag_mma(cur, acc);

  #pragma unroll
  for (int mi = 0; mi < 2; ++mi) {
    #pragma unroll
    for (int j = 0; j < 4; ++j) {
      const int n = n0 + j * 16 + lr;
      const float bb = bias ? bias[n] : 0.f;
      #pragma unroll
      for (int r = 0; r < 8; ++r) {
        const int m = m0 + mi * 16 + hs * 8 + r;  // C layout: VGPR r -> m0+r / m0+8+r
        float v = acc[mi][j][r] + bb;
        if (act == 1) v = 0.5f * v * (1.f + erff(v * 0.70710678f));
        size_t off = (size_t)m * N + n;
        if (resid) v += resid[off];
        C[off] = v;
      }
    }
  }
}

// ---------------------------------------------------------------------------
// TDM staging demonstrator: Tensor Data Mover loads a 128x128 f16 tile of
// W (row-major, K contiguous) into LDS, waits on TENSORcnt, then consumes
// it through ds_load into a WMMA.  D# layout per CDNA5 ISA.
// ---------------------------------------------------------------------------
__global__ __launch_bounds__(128)
void k_tdm_stage(const _Float16* __restrict__ W, float* __restrict__ out,
                 int K, int n0) {
  __shared__ _Float16 tile[128 * 128];

  unsigned long long ga = (unsigned long long)(const void*)(W + (size_t)n0 * K);
  u32x4 g0;
  g0[0] = 1u;                                           // count=1, user descriptor
  g0[1] = 0u;                                           // lds_addr (tile base)
  g0[2] = (unsigned)(ga & 0xffffffffu);                 // global_addr[31:0]
  g0[3] = (unsigned)((ga >> 32) & 0x01ffffffu) | (2u << 30); // addr[56:32] | type=2
  i32x8 g1;
  g1[0] = 0x00010000;                                   // data_size=1 (2 bytes)
  g1[1] = (int)((K & 0xffff) << 16);                    // tensor_dim0[15:0]
  g1[2] = (int)(((unsigned)K >> 16) | (128u << 16));    // tensor_dim0[31:16] | dim1[15:0]
  g1[3] = (int)(128u << 16);                            // tile_dim0 = 128
  g1[4] = 128;                                          // tile_dim1 = 128
  g1[5] = K;                                            // tensor_dim0_stride[31:0]
  g1[6] = 0;
  g1[7] = 0;
  i32x4 gz = {0, 0, 0, 0};
#if __has_include(<hip/amd_detail/amd_gfx1250_TDM.h>)
  i32x8 gz8 = {0, 0, 0, 0, 0, 0, 0, 0};
  __builtin_amdgcn_tensor_load_to_lds(g0, g1, gz, gz, gz8, 0);
#else
  __builtin_amdgcn_tensor_load_to_lds(g0, g1, gz, gz, 0);
#endif
  __builtin_amdgcn_s_wait_tensorcnt(0);
  __syncthreads();

  const int lane = threadIdx.x & 31;
  const int hs = lane >> 4, lr = lane & 15;
  // B fragment straight out of LDS (ds_load_b128 pairs)
  v16h bf = *(const v16h*)(&tile[(size_t)(lr + (threadIdx.x >> 5) * 16) * 128 + hs * 16]);
  v16h af = bf;
  v8f c = {};
  c = __builtin_amdgcn_wmma_f32_16x16x32_f16(false, af, false, bf, (short)0, c, false, false);
  if (__builtin_amdgcn_cluster_id_x() > 0) __builtin_amdgcn_s_cluster_barrier();
  #pragma unroll
  for (int r = 0; r < 8; ++r) out[(size_t)threadIdx.x * 8 + r] = c[r];
}

// ---------------------------------------------------------------------------
// Embedding: conv1d_circ + temporal PE from marks
// ---------------------------------------------------------------------------
__device__ __forceinline__ float pe_val(int pos, int o) {
  const int i = o >> 1;
  const float dv = __expf(-0.03597790f * (float)i);   // 2*ln(1e4)/512
  const float ph = (float)pos * dv;
  return (o & 1) ? __cosf(ph) : __sinf(ph);
}

__global__ void k_embed_enc(const float* __restrict__ x, const int* __restrict__ marks,
                            const float* __restrict__ Wc, float* __restrict__ out) {
  const int bl = blockIdx.x, b = bl >> 10, l = bl & 1023;
  __shared__ float xin[63];
  __shared__ int mk[4];
  if (threadIdx.x < 63) {
    int k = threadIdx.x / 21, c = threadIdx.x % 21;
    int ll = (l - 1 + k + LL) & (LL - 1);
    xin[k * 21 + c] = x[((size_t)(b << 10) + ll) * 21 + c];
  }
  if (threadIdx.x >= 64 && threadIdx.x < 68) mk[threadIdx.x - 64] = marks[(size_t)bl * 4 + threadIdx.x - 64];
  __syncthreads();
  for (int o = threadIdx.x; o < DM; o += 256) {
    float s = 0.f;
    const float* w = Wc + (size_t)o * 63;
    #pragma unroll
    for (int k = 0; k < 3; ++k)
      for (int c = 0; c < 21; ++c) s += xin[k * 21 + c] * w[c * 3 + k];
    float pe = 0.f;
    #pragma unroll
    for (int j = 0; j < 4; ++j) pe += pe_val(mk[j], o);
    out[(size_t)bl * DM + o] = s + pe;
  }
}

__global__ void k_embed_dec(const float* __restrict__ seas, const int* __restrict__ marks,
                            const float* __restrict__ Wc, float* __restrict__ out) {
  const int bl = blockIdx.x, b = bl >> 10, l = bl & 1023;
  __shared__ float xin[3];
  __shared__ int mk[4];
  if (threadIdx.x < 3) {
    int ll = (l - 1 + threadIdx.x + LL) & (LL - 1);
    xin[threadIdx.x] = seas[(size_t)(b << 10) + ll];
  }
  if (threadIdx.x >= 4 && threadIdx.x < 8) mk[threadIdx.x - 4] = marks[(size_t)bl * 4 + threadIdx.x - 4];
  __syncthreads();
  for (int o = threadIdx.x; o < DM; o += 256) {
    float s = xin[0] * Wc[o * 3] + xin[1] * Wc[o * 3 + 1] + xin[2] * Wc[o * 3 + 2];
    float pe = 0.f;
    #pragma unroll
    for (int j = 0; j < 4; ++j) pe += pe_val(mk[j], o);
    out[(size_t)bl * DM + o] = s + pe;
  }
}

// ---------------------------------------------------------------------------
// seasonal_init (channel 0) + trend0 init
// ---------------------------------------------------------------------------
__global__ void k_seasinit(const float* __restrict__ x, float* __restrict__ seas,
                           float* __restrict__ trend0) {
  const int b = blockIdx.x;
  __shared__ float red[256];
  __shared__ float meanv;
  float s = 0.f;
  for (int l = threadIdx.x; l < LL; l += 256) s += x[((size_t)(b << 10) + l) * 21];
  red[threadIdx.x] = s; __syncthreads();
  for (int st = 128; st > 0; st >>= 1) {
    if (threadIdx.x < st) red[threadIdx.x] += red[threadIdx.x + st];
    __syncthreads();
  }
  if (!threadIdx.x) meanv = red[0] * (1.f / (float)LL);
  __syncthreads();
  for (int l = threadIdx.x; l < LL; l += 256) {
    size_t o = (size_t)(b << 10) + l;
    if (l < 512) {
      const int src = 512 + l;
      float m = 0.f;
      #pragma unroll
      for (int j = -12; j <= 12; ++j) {
        int ll = src + j; ll = ll < 0 ? 0 : (ll > 1023 ? 1023 : ll);
        m += x[((size_t)(b << 10) + ll) * 21];
      }
      m *= (1.f / 25.f);
      seas[o] = x[((size_t)(b << 10) + src) * 21] - m;
      trend0[o] = m;
    } else {
      seas[o] = 0.f;
      trend0[o] = meanv;
    }
  }
}

// ---------------------------------------------------------------------------
// series_decomp: s_out = x - movavg25(x); optionally t_acc += movavg
// ---------------------------------------------------------------------------
__global__ void k_decomp(const float* __restrict__ x, float* __restrict__ s_out,
                         float* __restrict__ t_acc) {
  const int bl = blockIdx.x, b = bl >> 10, l = bl & 1023;
  const float* xb = x + ((size_t)b << 10) * DM;
  for (int c = threadIdx.x; c < DM; c += 256) {
    float s = 0.f;
    #pragma unroll
    for (int j = -12; j <= 12; ++j) {
      int ll = l + j; ll = ll < 0 ? 0 : (ll > 1023 ? 1023 : ll);
      s += xb[(size_t)ll * DM + c];
    }
    const float mm = s * (1.f / 25.f);
    const size_t idx = (size_t)bl * DM + c;
    s_out[idx] = x[idx] - mm;
    if (t_acc) t_acc[idx] += mm;
  }
}

// ---------------------------------------------------------------------------
// mean_corr[b,tau] = (1/512) * sum_l S[b, (l+tau)%L, l]
// ---------------------------------------------------------------------------
__global__ void k_diag(const float* __restrict__ S, float* __restrict__ corr) {
  const int tau = blockIdx.x, b = blockIdx.y;
  const float* Sb = S + (size_t)b * LL * LL;
  __shared__ float red[256];
  float s = 0.f;
  for (int l = threadIdx.x; l < LL; l += 256)
    s += Sb[(size_t)((l + tau) & (LL - 1)) * LL + l];
  red[threadIdx.x] = s; __syncthreads();
  for (int st = 128; st > 0; st >>= 1) {
    if (threadIdx.x < st) red[threadIdx.x] += red[threadIdx.x + st];
    __syncthreads();
  }
  if (!threadIdx.x) corr[(size_t)b * LL + tau] = red[0] * (1.f / 512.f);
}

// ---------------------------------------------------------------------------
// per-batch top-34 + softmax
// ---------------------------------------------------------------------------
__global__ void k_topk(const float* __restrict__ corr, float* __restrict__ wts,
                       int* __restrict__ dly) {
  const int b = blockIdx.x;
  __shared__ float v[LL];
  __shared__ float rv[256];
  __shared__ int   ri[256];
  __shared__ float selw[TOPK];
  __shared__ int   seld[TOPK];
  for (int l = threadIdx.x; l < LL; l += 256) v[l] = corr[(size_t)b * LL + l];
  __syncthreads();
  for (int it = 0; it < TOPK; ++it) {
    float bm = -3.0e38f; int bi = 0;
    for (int l = threadIdx.x; l < LL; l += 256)
      if (v[l] > bm) { bm = v[l]; bi = l; }
    rv[threadIdx.x] = bm; ri[threadIdx.x] = bi; __syncthreads();
    for (int st = 128; st > 0; st >>= 1) {
      if (threadIdx.x < st && rv[threadIdx.x + st] > rv[threadIdx.x]) {
        rv[threadIdx.x] = rv[threadIdx.x + st]; ri[threadIdx.x] = ri[threadIdx.x + st];
      }
      __syncthreads();
    }
    if (!threadIdx.x) { selw[it] = rv[0]; seld[it] = ri[0]; v[ri[0]] = -3.0e38f; }
    __syncthreads();
  }
  if (!threadIdx.x) {
    float mx = selw[0];
    for (int i = 1; i < TOPK; ++i) mx = fmaxf(mx, selw[i]);
    float ssum = 0.f;
    float e[TOPK];
    for (int i = 0; i < TOPK; ++i) { e[i] = __expf(selw[i] - mx); ssum += e[i]; }
    const float inv = 1.f / ssum;
    for (int i = 0; i < TOPK; ++i) {
      wts[b * TOPK + i] = e[i] * inv;
      dly[b * TOPK + i] = seld[i];
    }
  }
}

// ---------------------------------------------------------------------------
// agg[b,l,c] = sum_i w[b,i] * V[b,(l+delay[b,i])%L, c]
// ---------------------------------------------------------------------------
__global__ void k_agg(const float* __restrict__ V, const float* __restrict__ wts,
                      const int* __restrict__ dly, float* __restrict__ out) {
  const int bl = blockIdx.x, b = bl >> 10, l = bl & 1023;
  __shared__ float w[TOPK];
  __shared__ int   d[TOPK];
  if (threadIdx.x < TOPK) {
    w[threadIdx.x] = wts[b * TOPK + threadIdx.x];
    d[threadIdx.x] = dly[b * TOPK + threadIdx.x];
  }
  __syncthreads();
  for (int c = threadIdx.x; c < DM; c += 256) {
    float s = 0.f;
    #pragma unroll 4
    for (int i = 0; i < TOPK; ++i)
      s += w[i] * V[((size_t)(b << 10) + ((l + d[i]) & (LL - 1))) * DM + c];
    out[(size_t)bl * DM + c] = s;
  }
}

// ---------------------------------------------------------------------------
// layer_norm over channels, then column-mean subtraction (my_layernorm)
// ---------------------------------------------------------------------------
__global__ void k_ln(const float* __restrict__ x, const float* __restrict__ g,
                     const float* __restrict__ bta, float* __restrict__ y) {
  const int bl = blockIdx.x;
  const float* r = x + (size_t)bl * DM;
  __shared__ float rs[256], rq[256];
  __shared__ float mu, ri;
  float s = 0.f, q = 0.f;
  for (int c = threadIdx.x; c < DM; c += 256) { float v = r[c]; s += v; q += v * v; }
  rs[threadIdx.x] = s; rq[threadIdx.x] = q; __syncthreads();
  for (int st = 128; st > 0; st >>= 1) {
    if (threadIdx.x < st) { rs[threadIdx.x] += rs[threadIdx.x + st]; rq[threadIdx.x] += rq[threadIdx.x + st]; }
    __syncthreads();
  }
  if (!threadIdx.x) {
    mu = rs[0] * (1.f / (float)DM);
    float var = rq[0] * (1.f / (float)DM) - mu * mu;
    ri = rsqrtf(var + 1e-5f);
  }
  __syncthreads();
  for (int c = threadIdx.x; c < DM; c += 256)
    y[(size_t)bl * DM + c] = (r[c] - mu) * ri * g[c] + bta[c];
}

__global__ void k_colmean(const float* __restrict__ y, float* __restrict__ cm) {
  const int b = blockIdx.x, c = threadIdx.x;
  float s0 = 0.f, s1 = 0.f;
  for (int l = 0; l < LL; ++l) {
    const float* r = y + ((size_t)(b << 10) + l) * DM;
    s0 += r[c]; s1 += r[c + 256];
  }
  cm[b * DM + c]       = s0 * (1.f / (float)LL);
  cm[b * DM + c + 256] = s1 * (1.f / (float)LL);
}

__global__ void k_sub(float* __restrict__ y, const float* __restrict__ cm) {
  size_t i = (size_t)blockIdx.x * 256 + threadIdx.x;
  int c = (int)(i & (DM - 1));
  int b = (int)(i >> 19);            // 1024*512 = 2^19 elements per batch
  y[i] -= cm[b * DM + c];
}

// ---------------------------------------------------------------------------
// residual trend: trend0[b,l] += sum_{c,k} T[b,(l-1+k)%L,c] * wt[c*3+k]
// ---------------------------------------------------------------------------
__global__ void k_trendconv(const float* __restrict__ T, const float* __restrict__ wt,
                            float* __restrict__ trend0) {
  const int bl = blockIdx.x, b = bl >> 10, l = bl & 1023;
  __shared__ float red[256];
  float s = 0.f;
  for (int t = threadIdx.x; t < DM * 3; t += 256) {
    const int c = t / 3, k = t - 3 * c;
    const int ll = (l - 1 + k + LL) & (LL - 1);
    s += T[((size_t)(b << 10) + ll) * DM + c] * wt[t];
  }
  red[threadIdx.x] = s; __syncthreads();
  for (int st = 128; st > 0; st >>= 1) {
    if (threadIdx.x < st) red[threadIdx.x] += red[threadIdx.x + st];
    __syncthreads();
  }
  if (!threadIdx.x) trend0[bl] += red[0];
}

// ---------------------------------------------------------------------------
// final: out[b,l] = trend0 + dot(decLN, proj_w) + proj_b
// ---------------------------------------------------------------------------
__global__ void k_final(const float* __restrict__ y, const float* __restrict__ pw,
                        const float* __restrict__ pb, const float* __restrict__ trend0,
                        float* __restrict__ out) {
  const int bl = blockIdx.x;
  __shared__ float red[256];
  float s = 0.f;
  for (int c = threadIdx.x; c < DM; c += 256) s += y[(size_t)bl * DM + c] * pw[c];
  red[threadIdx.x] = s; __syncthreads();
  for (int st = 128; st > 0; st >>= 1) {
    if (threadIdx.x < st) red[threadIdx.x] += red[threadIdx.x + st];
    __syncthreads();
  }
  if (!threadIdx.x) out[bl] = trend0[bl] + red[0] + pb[0];
}

// ===========================================================================
// Host orchestration
// ===========================================================================
extern "C" void kernel_launch(void* const* d_in, const int* in_sizes, int n_in,
                              void* d_out, int out_size, void* d_ws, size_t ws_size,
                              hipStream_t stream) {
  (void)in_sizes; (void)n_in; (void)out_size; (void)ws_size;
  const float* x_enc = (const float*)d_in[0];
  const int*   mk_e  = (const int*)d_in[1];
  const int*   mk_d  = (const int*)d_in[3];
  auto P = [&](int i) { return (const float*)d_in[i]; };

  // ---- workspace bump allocator ----
  char* base = (char*)d_ws;
  size_t off = 0;
  auto alloc = [&](size_t bytes) -> void* {
    off = (off + 255) & ~(size_t)255;
    void* p = base + off;
    off += bytes;
    return p;
  };
  const size_t MD  = (size_t)MROWS * DM;     // 8.4M elems
  const size_t MDF = (size_t)MROWS * DFF;    // 33.6M elems
  float* X   = (float*)alloc(MD * 4);
  float* Y   = (float*)alloc(MD * 4);
  float* Qb  = (float*)alloc(MD * 4);
  float* Kb  = (float*)alloc(MD * 4);
  float* Vb  = (float*)alloc(MD * 4);
  float* ENC = (float*)alloc(MD * 4);
  float* T   = (float*)alloc(MD * 4);
  float* H   = (float*)alloc(MDF * 4);
  float* Sb  = (float*)alloc((size_t)BB * LL * LL * 4);
  float* corr   = (float*)alloc((size_t)BB * LL * 4);
  float* wts    = (float*)alloc((size_t)BB * TOPK * 4);
  int*   dly    = (int*)  alloc((size_t)BB * TOPK * 4);
  float* trend0 = (float*)alloc((size_t)BB * LL * 4);
  float* seas   = (float*)alloc((size_t)BB * LL * 4);
  float* CM     = (float*)alloc((size_t)BB * DM * 4);
  _Float16* A16 = (_Float16*)alloc(MDF * 2);
  _Float16* Q16 = (_Float16*)alloc(MD * 2);
  _Float16* K16 = (_Float16*)alloc(MD * 2);

  auto cvt = [&](const float* s, _Float16* d, size_t n) {
    k_cvt<<<(unsigned)((n + 255) / 256), 256, 0, stream>>>(s, d, n);
  };
  auto cvtW = [&](int idx, size_t n) -> const _Float16* {
    _Float16* d = (_Float16*)alloc(n * 2);
    cvt(P(idx), d, n);
    return d;
  };
  auto gemm = [&](const _Float16* A, const _Float16* W, const float* bias,
                  const float* resid, float* C, int M, int N, int K, int act,
                  long long sA, long long sW, long long sC, int batch) {
    dim3 g(N / 128, M / 64, batch);
    k_gemm<<<g, 128, 0, stream>>>(A, W, bias, resid, C, M, N, K, act, sA, sW, sC);
  };

  // ---- convert all GEMM weights to f16 once ----
  const _Float16* encW[2][6];
  for (int i = 0; i < 2; ++i) {
    int b6 = 6 + i * 10;
    encW[i][0] = cvtW(b6 + 0, (size_t)DM * DM);   // wq
    encW[i][1] = cvtW(b6 + 2, (size_t)DM * DM);   // wk
    encW[i][2] = cvtW(b6 + 4, (size_t)DM * DM);   // wv
    encW[i][3] = cvtW(b6 + 6, (size_t)DM * DM);   // wo
    encW[i][4] = cvtW(b6 + 8, (size_t)DFF * DM);  // w1
    encW[i][5] = cvtW(b6 + 9, (size_t)DM * DFF);  // w2
  }
  const _Float16* decW[2][10];
  for (int i = 0; i < 2; ++i) {
    int bb = 28 + i * 19;
    decW[i][0] = cvtW(bb + 0,  (size_t)DM * DM);  // self wq
    decW[i][1] = cvtW(bb + 2,  (size_t)DM * DM);
    decW[i][2] = cvtW(bb + 4,  (size_t)DM * DM);
    decW[i][3] = cvtW(bb + 6,  (size_t)DM * DM);
    decW[i][4] = cvtW(bb + 8,  (size_t)DM * DM);  // cross wq
    decW[i][5] = cvtW(bb + 10, (size_t)DM * DM);
    decW[i][6] = cvtW(bb + 12, (size_t)DM * DM);
    decW[i][7] = cvtW(bb + 14, (size_t)DM * DM);
    decW[i][8] = cvtW(bb + 16, (size_t)DFF * DM); // w1
    decW[i][9] = cvtW(bb + 17, (size_t)DM * DFF); // w2
  }

  // ---- autocorrelation attention: out = q_in + AC(q,k,v) @ wo^T + bo ----
  auto attn = [&](const float* qin, const float* kin,
                  const _Float16* wq, const float* bq,
                  const _Float16* wk, const float* bk,
                  const _Float16* wv, const float* bv,
                  const _Float16* wo, const float* bo, float* outb) {
    cvt(kin, A16, MD);
    gemm(A16, wk, bk, nullptr, Kb, MROWS, DM, DM, 0, 0, 0, 0, 1);
    gemm(A16, wv, bv, nullptr, Vb, MROWS, DM, DM, 0, 0, 0, 0, 1);
    cvt(qin, A16, MD);
    gemm(A16, wq, bq, nullptr, Qb, MROWS, DM, DM, 0, 0, 0, 0, 1);
    cvt(Qb, Q16, MD);
    cvt(Kb, K16, MD);
    // S[b] = Q[b] K[b]^T  (exact replacement of the FFT correlation)
    gemm(Q16, K16, nullptr, nullptr, Sb, LL, LL, DM, 0,
         (long long)LL * DM, (long long)LL * DM, (long long)LL * LL, BB);
    k_diag<<<dim3(LL, BB), 256, 0, stream>>>(Sb, corr);
    k_topk<<<BB, 256, 0, stream>>>(corr, wts, dly);
    k_agg<<<BB * LL, 256, 0, stream>>>(Vb, wts, dly, outb);
    cvt(outb, A16, MD);
    gemm(A16, wo, bo, qin, outb, MROWS, DM, DM, 0, 0, 0, 0, 1);
  };
  auto ffn = [&](const float* xin, const _Float16* w1, const _Float16* w2, float* outb) {
    cvt(xin, A16, MD);
    gemm(A16, w1, nullptr, nullptr, H, MROWS, DFF, DM, 1, 0, 0, 0, 1);  // GELU
    cvt(H, A16, MDF);
    gemm(A16, w2, nullptr, xin, outb, MROWS, DM, DFF, 0, 0, 0, 0, 1);
  };
  auto mln = [&](const float* xin, const float* g, const float* b, float* outb) {
    k_ln<<<MROWS, 256, 0, stream>>>(xin, g, b, outb);
    k_colmean<<<BB, 256, 0, stream>>>(outb, CM);
    k_sub<<<(unsigned)(MD / 256), 256, 0, stream>>>(outb, CM);
  };

  // ================= encoder =================
  k_embed_enc<<<MROWS, 256, 0, stream>>>(x_enc, mk_e, P(4), X);
  for (int i = 0; i < 2; ++i) {
    int b6 = 6 + i * 10;
    attn(X, X, encW[i][0], P(b6 + 1), encW[i][1], P(b6 + 3),
               encW[i][2], P(b6 + 5), encW[i][3], P(b6 + 7), Y);
    k_decomp<<<MROWS, 256, 0, stream>>>(Y, X, nullptr);
    ffn(X, encW[i][4], encW[i][5], Y);
    k_decomp<<<MROWS, 256, 0, stream>>>(Y, X, nullptr);
  }
  mln(X, P(26), P(27), ENC);

  // ================= decoder =================
  k_seasinit<<<BB, 256, 0, stream>>>(x_enc, seas, trend0);
  k_embed_dec<<<MROWS, 256, 0, stream>>>(seas, mk_d, P(5), X);
  for (int i = 0; i < 2; ++i) {
    int bb = 28 + i * 19;
    k_fill<<<(unsigned)(MD / 256), 256, 0, stream>>>(T, MD);
    attn(X, X, decW[i][0], P(bb + 1), decW[i][1], P(bb + 3),
               decW[i][2], P(bb + 5), decW[i][3], P(bb + 7), Y);
    k_decomp<<<MROWS, 256, 0, stream>>>(Y, X, T);
    attn(X, ENC, decW[i][4], P(bb + 9), decW[i][5], P(bb + 11),
                 decW[i][6], P(bb + 13), decW[i][7], P(bb + 15), Y);
    k_decomp<<<MROWS, 256, 0, stream>>>(Y, X, T);
    ffn(X, decW[i][8], decW[i][9], Y);
    k_decomp<<<MROWS, 256, 0, stream>>>(Y, X, T);
    k_trendconv<<<MROWS, 256, 0, stream>>>(T, P(bb + 18), trend0);
  }
  mln(X, P(66), P(67), Y);
  k_final<<<MROWS, 256, 0, stream>>>(Y, P(68), P(69), trend0, (float*)d_out);
}
